// NeuralODEClassifier_36799279792354
// MI455X (gfx1250) — matrix operand order repaired and checked by reading, not compiled, not run on hardware
//
#include <hip/hip_runtime.h>
#include <stdint.h>

// ---------------- problem constants ----------------
#define BATCH 2048
#define DIM   256
#define HID   512
#define NCLS  10
#define TPTS  8
#define SUBSTEPS 4
#define NSTEPS ((TPTS - 1) * SUBSTEPS)   // 28 dopri5 steps

// ---------------- tiling ----------------
#define MTILE    16          // batch rows per workgroup (WMMA M)
#define NTHREADS 256         // 8 wave32 waves
#define NWAVES   8

// LDS row strides (bf16 elements). Chosen odd-dword-ish so the 16 rows of an
// A-fragment read map to distinct LDS banks (stride/2 dwords ≡ 9 mod 64).
#define SA  274              // Ybf stride  (DIM=256 + pad)
#define SH  530              // Hbf stride  (HID=512 + pad)
#define SYF 264              // final-y fp32 stride
#define SHH 72               // hidden head fp32 stride

typedef __attribute__((ext_vector_type(16))) __bf16   v16bf;
typedef __attribute__((ext_vector_type(8)))  float    v8f;
typedef uint32_t u32x4 __attribute__((ext_vector_type(4)));

union FragBf {
  v16bf    v;
  uint32_t u[8];
  u32x4    q[2];
};

__device__ __forceinline__ uint16_t f32_to_bf16(float x) {
  uint32_t u = __float_as_uint(x);
  u += 0x7FFFu + ((u >> 16) & 1u);   // round-to-nearest-even
  return (uint16_t)(u >> 16);
}

// -----------------------------------------------------------------------------
// Repack a row-major fp32 weight matrix W[K][N] into bf16 WMMA-B fragments.
// Fragment (kb, nb) covers K rows [kb*32, kb*32+32) x N cols [nb*16, nb*16+16),
// stored contiguously (512 bf16 = 1KB) in per-lane order so a wave loads its
// fragment with two coalesced global_load_b128 per lane.
// B layout (32x16, 16-bit): lane L -> column L%16; lanes 0-15 hold K=0..7,16..23,
// lanes 16-31 hold K=8..15,24..31; VGPR v holds pair (K=8h+2v, +1), +8 more if v>=4 (h=L/16).
// -----------------------------------------------------------------------------
__global__ void repack_weights_kernel(const float* __restrict__ W,
                                      uint16_t* __restrict__ dst,
                                      int K, int N) {
  int e = blockIdx.x * blockDim.x + threadIdx.x;
  if (e >= K * N) return;
  int frag = e >> 9;                // /512
  int r    = e & 511;
  int lane = r >> 4;                // 0..31
  int q    = r & 15;                // element within lane (16 bf16 per lane)
  int v    = q >> 1;                // packed dword index 0..7
  int p    = q & 1;
  int NB   = N >> 4;
  int kb   = frag / NB;
  int nb   = frag - kb * NB;
  int kk   = kb * 32 + ((lane >= 16) ? 8 : 0) + ((v & 4) ? 16 : 0) + (v & 3) * 2 + p;
  int nn   = nb * 16 + (lane & 15);
  dst[e] = f32_to_bf16(W[(size_t)kk * N + nn]);
}

// -----------------------------------------------------------------------------
// Accumulate NT adjacent 16x16 output tiles: acc += A_lds @ Bpack over K=KB*32.
// A layout (16x32, 16-bit, ISA 7.12.2): lane L -> row L%16; for lanes 0-15
// VGPR v holds K pairs {(0,1),(2,3),(4,5),(6,7),(16,17),(18,19),(20,21),(22,23)}
// and lanes 16-31 the same +8.
// -----------------------------------------------------------------------------
template <int KB, int NT>
__device__ __forceinline__ void mma_block(const uint16_t* lds, const int stride,
                                          const uint16_t* __restrict__ bp,
                                          const int nb0, const int NBtot,
                                          const int lane, v8f acc[NT]) {
  const int row_l = lane & 15;
  const int hi    = lane >> 4;
#pragma unroll
  for (int kb = 0; kb < KB; ++kb) {
    FragBf a;
    const int kbase = kb * 32 + hi * 8;
#pragma unroll
    for (int v = 0; v < 8; ++v) {
      const int kk = kbase + ((v & 4) ? 16 : 0) + (v & 3) * 2;
      a.u[v] = *(const uint32_t*)(lds + row_l * stride + kk);   // ds_load_b32
    }
#pragma unroll
    for (int tt = 0; tt < NT; ++tt) {
      FragBf b;
      const uint16_t* src = bp + ((size_t)(kb * NBtot + nb0 + tt) << 9) + lane * 16;
      b.q[0] = *(const u32x4*)(src);          // global_load_b128
      b.q[1] = *(const u32x4*)(src + 8);      // global_load_b128
      acc[tt] = __builtin_amdgcn_wmma_f32_16x16x32_bf16(
          false, a.v, false, b.v, (short)0, acc[tt], false, false);
    }
  }
}

template <int N> struct IC { static constexpr int value = N; };

// -----------------------------------------------------------------------------
// Persistent Neural-ODE kernel: one workgroup integrates 16 batch rows through
// all 28 dopri5 steps (each row's trajectory is independent), then applies the
// classifier head. State y / k1..k5 / y_next live in fp32 registers,
// column-partitioned across the 8 waves; bf16 activations staged through LDS.
// -----------------------------------------------------------------------------
__global__ __launch_bounds__(NTHREADS, 1)
void node_dopri5_kernel(const float* __restrict__ x0,
                        const float* __restrict__ tgrid,
                        const float* __restrict__ b1g,
                        const float* __restrict__ b2g,
                        const float* __restrict__ Wc1,
                        const float* __restrict__ bc1,
                        const float* __restrict__ Wc2,
                        const float* __restrict__ bc2,
                        const uint16_t* __restrict__ W1p,
                        const uint16_t* __restrict__ W2p,
                        float* __restrict__ out) {
  __shared__ __align__(16) uint16_t Ybf[MTILE * SA];   // bf16 ytmp (A of GEMM1)
  __shared__ __align__(16) uint16_t Hbf[MTILE * SH];   // bf16 tanh(h) (A of GEMM2)
  __shared__ __align__(16) float    Yf[MTILE * SYF];   // fp32 final y (head)
  __shared__ __align__(16) float    Hh[MTILE * SHH];   // fp32 relu hidden (head)

  const int tid   = threadIdx.x;
  const int wave  = tid >> 5;
  const int lane  = tid & 31;
  const int row_l = lane & 15;
  const int hi    = lane >> 4;
  const int wg    = blockIdx.x;
  const int row0  = wg * MTILE;

  // C/D fragment layout (f32 16x16): lane L, VGPR r -> (row = r + 8*hi, col = L%16).
  float y[2][8];
#pragma unroll
  for (int tle = 0; tle < 2; ++tle) {
    const int col = (wave * 2 + tle) * 16 + row_l;
#pragma unroll
    for (int r = 0; r < 8; ++r)
      y[tle][r] = x0[(size_t)(row0 + r + hi * 8) * DIM + col];
  }

  // Biases for this wave's output columns (loop-invariant).
  float biasA[4], biasB[2];
#pragma unroll
  for (int tt = 0; tt < 4; ++tt) biasA[tt] = b1g[(wave * 4 + tt) * 16 + row_l];
#pragma unroll
  for (int tt = 0; tt < 2; ++tt) biasB[tt] = b2g[(wave * 2 + tt) * 16 + row_l];

  float k[5][2][8];   // k1..k5 (k6 fused into y_next directly)
  float yn[2][8];     // y_next accumulator

  // Dormand-Prince tableau
  constexpr float AC[6][5] = {
      {0, 0, 0, 0, 0},
      {1.0f / 5.0f, 0, 0, 0, 0},
      {3.0f / 40.0f, 9.0f / 40.0f, 0, 0, 0},
      {44.0f / 45.0f, -56.0f / 15.0f, 32.0f / 9.0f, 0, 0},
      {19372.0f / 6561.0f, -25360.0f / 2187.0f, 64448.0f / 6561.0f,
       -212.0f / 729.0f, 0},
      {9017.0f / 3168.0f, -355.0f / 33.0f, 46732.0f / 5247.0f, 49.0f / 176.0f,
       -5103.0f / 18656.0f}};
  constexpr float BC[6] = {35.0f / 384.0f,   0.0f,
                           500.0f / 1113.0f, 125.0f / 192.0f,
                           -2187.0f / 6784.0f, 11.0f / 84.0f};

  for (int s = 0; s < NSTEPS; ++s) {
    const int   iv = s >> 2;
    const float dt = (tgrid[iv + 1] - tgrid[iv]) * (1.0f / SUBSTEPS);

    auto stage_fn = [&](auto SC) {
      constexpr int stage = decltype(SC)::value;

      // ---- ytmp = y + dt * sum_j a[stage][j] * k_j  ->  Ybf (bf16) ----
#pragma unroll
      for (int tle = 0; tle < 2; ++tle) {
        const int col = (wave * 2 + tle) * 16 + row_l;
#pragma unroll
        for (int r = 0; r < 8; ++r) {
          float v = y[tle][r];
          if constexpr (stage >= 1) v += dt * AC[stage][0] * k[0][tle][r];
          if constexpr (stage >= 2) v += dt * AC[stage][1] * k[1][tle][r];
          if constexpr (stage >= 3) v += dt * AC[stage][2] * k[2][tle][r];
          if constexpr (stage >= 4) v += dt * AC[stage][3] * k[3][tle][r];
          if constexpr (stage >= 5) v += dt * AC[stage][4] * k[4][tle][r];
          Ybf[(r + hi * 8) * SA + col] = f32_to_bf16(v);
        }
      }
      __syncthreads();

      // ---- GEMM1: H = tanh(ytmp @ W1 + b1) -> Hbf (bf16) ----
      v8f accA[4] = {};
      mma_block<DIM / 32, 4>(Ybf, SA, W1p, wave * 4, HID / 16, lane, accA);
#pragma unroll
      for (int tt = 0; tt < 4; ++tt) {
        const int col = (wave * 4 + tt) * 16 + row_l;
#pragma unroll
        for (int r = 0; r < 8; ++r)
          Hbf[(r + hi * 8) * SH + col] = f32_to_bf16(tanhf(accA[tt][r] + biasA[tt]));
      }
      __syncthreads();

      // ---- GEMM2: k_stage = H @ W2 + b2 (stays in registers) ----
      v8f accB[2] = {};
      mma_block<HID / 32, 2>(Hbf, SH, W2p, wave * 2, DIM / 16, lane, accB);
#pragma unroll
      for (int tt = 0; tt < 2; ++tt) {
#pragma unroll
        for (int r = 0; r < 8; ++r) {
          const float kv = accB[tt][r] + biasB[tt];
          if constexpr (stage < 5) k[stage][tt][r] = kv;
          if constexpr (stage == 0)
            yn[tt][r] = y[tt][r] + dt * BC[0] * kv;
          else if constexpr (BC[stage] != 0.0f)
            yn[tt][r] += dt * BC[stage] * kv;
        }
      }
    };
    stage_fn(IC<0>{});
    stage_fn(IC<1>{});
    stage_fn(IC<2>{});
    stage_fn(IC<3>{});
    stage_fn(IC<4>{});
    stage_fn(IC<5>{});

#pragma unroll
    for (int tle = 0; tle < 2; ++tle)
#pragma unroll
      for (int r = 0; r < 8; ++r) y[tle][r] = yn[tle][r];
  }

  // ---------------- classifier head (tiny; plain fp32 FMAs) ----------------
#pragma unroll
  for (int tle = 0; tle < 2; ++tle) {
    const int col = (wave * 2 + tle) * 16 + row_l;
#pragma unroll
    for (int r = 0; r < 8; ++r)
      Yf[(r + hi * 8) * SYF + col] = y[tle][r];
  }
  __syncthreads();

  for (int e = tid; e < MTILE * 64; e += NTHREADS) {
    const int rr = e >> 6;
    const int cc = e & 63;
    float acc = bc1[cc];
    for (int kx = 0; kx < DIM; ++kx)
      acc = fmaf(Yf[rr * SYF + kx], Wc1[(size_t)kx * 64 + cc], acc);
    Hh[rr * SHH + cc] = fmaxf(acc, 0.0f);
  }
  __syncthreads();

  for (int e = tid; e < MTILE * NCLS; e += NTHREADS) {
    const int rr = e / NCLS;
    const int cc = e - rr * NCLS;
    float acc = bc2[cc];
    for (int kx = 0; kx < 64; ++kx)
      acc = fmaf(Hh[rr * SHH + kx], Wc2[(size_t)kx * NCLS + cc], acc);
    out[(size_t)(row0 + rr) * NCLS + cc] = acc;
  }
}

// -----------------------------------------------------------------------------
extern "C" void kernel_launch(void* const* d_in, const int* in_sizes, int n_in,
                              void* d_out, int out_size, void* d_ws,
                              size_t ws_size, hipStream_t stream) {
  (void)in_sizes; (void)n_in; (void)out_size; (void)ws_size;

  const float* x0  = (const float*)d_in[0];
  const float* t   = (const float*)d_in[1];
  const float* W1  = (const float*)d_in[2];
  const float* b1  = (const float*)d_in[3];
  const float* W2  = (const float*)d_in[4];
  const float* b2  = (const float*)d_in[5];
  const float* Wc1 = (const float*)d_in[6];
  const float* bc1 = (const float*)d_in[7];
  const float* Wc2 = (const float*)d_in[8];
  const float* bc2 = (const float*)d_in[9];
  float* out = (float*)d_out;

  uint16_t* W1p = (uint16_t*)d_ws;                    // 256 KB bf16 fragments
  uint16_t* W2p = W1p + (size_t)DIM * HID;            // 256 KB bf16 fragments

  repack_weights_kernel<<<(DIM * HID + 255) / 256, 256, 0, stream>>>(W1, W1p, DIM, HID);
  repack_weights_kernel<<<(HID * DIM + 255) / 256, 256, 0, stream>>>(W2, W2p, HID, DIM);

  node_dopri5_kernel<<<BATCH / MTILE, NTHREADS, 0, stream>>>(
      x0, t, b1, b2, Wc1, bc1, Wc2, bc2, W1p, W2p, out);
}
